// TensorProductConvLayer_42528766165474
// MI455X (gfx1250) — matrix operand order repaired and checked by reading, not compiled, not run on hardware
//
#include <hip/hip_runtime.h>

typedef __attribute__((ext_vector_type(16))) _Float16 v16h;
typedef __attribute__((ext_vector_type(8)))  float    v8f;

#define TILE_E 16
#define NBLK   256

#define INV_SQRT_FAN 0.14433756729740643f  /* 1/sqrt(48) */
#define INV_SQRT3    0.57735026918962576f  /* 1/sqrt(3)  */

// index (in halves) of element (row e, col k) inside the A-matrix (16x32 f16)
// fragment array laid out as [kk][lane][16 halves], kk = k>>5.
__device__ __forceinline__ int a_sw_idx(int e, int k) {
  int kk = k >> 5;
  int K  = k & 31;
  int ln = (e & 15) | (((K >> 3) & 1) << 4);
  int j  = (K & 7) | (((K >> 4) & 1) << 3);
  return (kk * 32 + ln) * 16 + j;
}

// ---------------------------------------------------------------------------
// Pre-swizzle W1 (128x128) and W2 (128x2304) from f32 row-major [K][N] into
// f16 B-matrix (32x16) fragment layout: [nt][kk][lane][16 halves], where
// lane: col = lane&15, kbase = 8*(lane>>4); half j -> K = kk*32 + kbase + (j&7) + 16*(j>>3).
// ---------------------------------------------------------------------------
__global__ void swizzle_weights(const float* __restrict__ W1,
                                const float* __restrict__ W2,
                                _Float16* __restrict__ W1h,
                                _Float16* __restrict__ W2h) {
  int tid = blockIdx.x * blockDim.x + threadIdx.x;
  const int W1H = 8 * 4 * 512;       // 16384 halves
  const int W2H = 144 * 4 * 512;     // 294912 halves
  if (tid < W1H) {
    int j = tid & 15, lane = (tid >> 4) & 31, kf = tid >> 9;
    int kk = kf & 3, nt = kf >> 2;
    int k = kk * 32 + ((lane >> 4) << 3) + (j & 7) + ((j >> 3) << 4);
    int n = nt * 16 + (lane & 15);
    W1h[tid] = (_Float16)W1[k * 128 + n];
  } else if (tid < W1H + W2H) {
    int t = tid - W1H;
    int j = t & 15, lane = (t >> 4) & 31, kf = t >> 9;
    int kk = kf & 3, nt = kf >> 2;
    int k = kk * 32 + ((lane >> 4) << 3) + (j & 7) + ((j >> 3) << 4);
    int n = nt * 16 + (lane & 15);
    W2h[t] = (_Float16)W2[k * 2304 + n];
  }
}

// ---------------------------------------------------------------------------
// Fused: LayerNorm -> GEMM1(relu) -> GEMM2 -> tensor product, 16 edges/block.
// ---------------------------------------------------------------------------
__launch_bounds__(NBLK)
__global__ void tpconv_fused(const float* __restrict__ attr1,     // E x 80
                             const float* __restrict__ attr2,     // E x 4
                             const float* __restrict__ edge_attr, // E x 128
                             const float* __restrict__ ln_g,
                             const float* __restrict__ ln_b,
                             const _Float16* __restrict__ W1h,
                             const float* __restrict__ b1,
                             const _Float16* __restrict__ W2h,
                             const float* __restrict__ b2,
                             float* __restrict__ out) {           // E x 80
  __shared__ __align__(32) _Float16 hln_sw[2048]; // A-frags of LN output
  __shared__ __align__(32) _Float16 h1_sw[2048];  // A-frags of relu(h@W1+b1)
  __shared__ float s1_arr[16][32];
  __shared__ float s1s2[16][32];
  __shared__ float dvv3[16][16];
  __shared__ float v1s2[16][16][3];
  __shared__ float v2l[16][3];
  __shared__ float outs_acc[16][32];
  __shared__ float sv_acc[16][16];
  __shared__ float vs_acc[16][16][3];

  const int tid  = threadIdx.x;
  const int lane = tid & 31;
  const int wave = tid >> 5;
  const int eb   = blockIdx.x * TILE_E;

  // ---- Phase A: LayerNorm (wave -> 2 edges, 16 lanes/edge, 8 elems/lane) ----
  {
    int e = wave * 2 + (lane >> 4);
    int s = lane & 15;
    const float* row = edge_attr + (size_t)(eb + e) * 128;
    float v[8];
    float sum = 0.f, sq = 0.f;
#pragma unroll
    for (int i = 0; i < 8; ++i) {
      v[i] = row[s + 16 * i];
      sum += v[i];
      sq  += v[i] * v[i];
    }
#pragma unroll
    for (int m = 1; m < 16; m <<= 1) {
      sum += __shfl_xor(sum, m, 16);
      sq  += __shfl_xor(sq,  m, 16);
    }
    float mu   = sum * (1.f / 128.f);
    float var  = sq * (1.f / 128.f) - mu * mu;
    float rstd = rsqrtf(var + 1e-5f);
#pragma unroll
    for (int i = 0; i < 8; ++i) {
      int k = s + 16 * i;
      float h = (v[i] - mu) * rstd * ln_g[k] + ln_b[k];
      hln_sw[a_sw_idx(e, k)] = (_Float16)h;
    }
  }
  // ---- Phase A: tensor-product coefficients + zero accumulators ----
  for (int idx = tid; idx < 16 * 32; idx += NBLK) {
    int e = idx >> 5, u = idx & 31;
    float s1v = attr1[(size_t)(eb + e) * 80 + u];
    float s2v = attr2[(size_t)(eb + e) * 4];
    s1_arr[e][u]   = s1v;
    s1s2[e][u]     = s1v * s2v;
    outs_acc[e][u] = 0.f;
  }
  for (int idx = tid; idx < 16 * 16; idx += NBLK) {
    int e = idx >> 4, u = idx & 15;
    float s2v = attr2[(size_t)(eb + e) * 4];
    float d = 0.f;
#pragma unroll
    for (int c = 0; c < 3; ++c) {
      float v1c = attr1[(size_t)(eb + e) * 80 + 32 + u * 3 + c];
      float v2c = attr2[(size_t)(eb + e) * 4 + 1 + c];
      v1s2[e][u][c]   = v1c * s2v;
      vs_acc[e][u][c] = 0.f;
      d += v1c * v2c;
    }
    dvv3[e][u]  = d * INV_SQRT3;
    sv_acc[e][u] = 0.f;
  }
  for (int idx = tid; idx < 48; idx += NBLK) {
    int e = idx / 3, c = idx - 3 * e;
    v2l[e][c] = attr2[(size_t)(eb + e) * 4 + 1 + c];
  }
  __syncthreads();

  // ---- GEMM1: h1 = relu(hln @ W1 + b1), one 16x16 N-tile per wave ----
  {
    const int nt = wave;  // 0..7
    v8f acc = {0.f, 0.f, 0.f, 0.f, 0.f, 0.f, 0.f, 0.f};
#pragma unroll
    for (int kk = 0; kk < 4; ++kk) {
      v16h a = *(const v16h*)&hln_sw[(kk * 32 + lane) * 16];
      v16h b = *(const v16h*)&W1h[((nt * 4 + kk) * 32 + lane) * 16];
      acc = __builtin_amdgcn_wmma_f32_16x16x32_f16(
          false, a, false, b, (short)0, acc, false, false);
    }
    const int kcol = nt * 16 + (lane & 15);
    const float bias = b1[kcol];
    const int ehi = (lane >> 4) * 8;
#pragma unroll
    for (int r = 0; r < 8; ++r) {
      float vv = acc[r] + bias;
      vv = vv > 0.f ? vv : 0.f;
      h1_sw[a_sw_idx(ehi + r, kcol)] = (_Float16)vv;
    }
  }
  __syncthreads();

  // ---- GEMM2 + immediate tensor-product contraction (w never hits memory) --
  // Scatter one 16x16 tile of w = h1@W2 + b2 into the LDS accumulators.
  const int col16 = lane & 15;
  const int ehi   = (lane >> 4) * 8;
  auto scatter = [&](int nt, const v8f& acc) {
    const float bias = b2[nt * 16 + col16];
    if (nt < 64) {                       // w_ss block: col = u*32 + w
      int u = nt >> 1;
      int w = ((nt & 1) << 4) | col16;
#pragma unroll
      for (int r = 0; r < 8; ++r) {
        int e = ehi + r;
        atomicAdd(&outs_acc[e][w], s1s2[e][u] * (acc[r] + bias));
      }
    } else if (nt < 96) {                // w_sv block: col-1024 = u*16 + w
      int u = nt - 64;
#pragma unroll
      for (int r = 0; r < 8; ++r) {
        int e = ehi + r;
        atomicAdd(&sv_acc[e][col16], s1_arr[e][u] * (acc[r] + bias));
      }
    } else if (nt < 112) {               // w_vs block: col-1536 = u*16 + w
      int u = nt - 96;
#pragma unroll
      for (int r = 0; r < 8; ++r) {
        int e = ehi + r;
        float vv = acc[r] + bias;
        atomicAdd(&vs_acc[e][col16][0], v1s2[e][u][0] * vv);
        atomicAdd(&vs_acc[e][col16][1], v1s2[e][u][1] * vv);
        atomicAdd(&vs_acc[e][col16][2], v1s2[e][u][2] * vv);
      }
    } else {                             // w_vv block: col-1792 = u*32 + w
      int t = nt - 112;
      int u = t >> 1;
      int w = ((t & 1) << 4) | col16;
#pragma unroll
      for (int r = 0; r < 8; ++r) {
        int e = ehi + r;
        atomicAdd(&outs_acc[e][w], dvv3[e][u] * (acc[r] + bias));
      }
    }
  };

  {
    // Hoist the 4 A-fragments (h1) into registers: invariant over all tiles.
    v16h afr[4];
#pragma unroll
    for (int kk = 0; kk < 4; ++kk)
      afr[kk] = *(const v16h*)&h1_sw[(kk * 32 + lane) * 16];

    const v16h* Wv = (const v16h*)W2h;   // frag (nt,kk,lane) at nt*128+kk*32+lane
    const int base = wave * 18;          // contiguous 18 tiles per wave

#pragma unroll 1
    for (int i = 0; i < 18; i += 2) {
      const int nt0 = base + i;
      const v16h* Bp = Wv + (size_t)nt0 * 128 + lane;
      v8f acc0 = {0.f, 0.f, 0.f, 0.f, 0.f, 0.f, 0.f, 0.f};
      v8f acc1 = {0.f, 0.f, 0.f, 0.f, 0.f, 0.f, 0.f, 0.f};
#pragma unroll
      for (int kk = 0; kk < 4; ++kk) {
        v16h b0 = Bp[kk * 32];           // tile nt0
        v16h b1f = Bp[128 + kk * 32];    // tile nt0+1
        acc0 = __builtin_amdgcn_wmma_f32_16x16x32_f16(
            false, afr[kk], false, b0, (short)0, acc0, false, false);
        acc1 = __builtin_amdgcn_wmma_f32_16x16x32_f16(
            false, afr[kk], false, b1f, (short)0, acc1, false, false);
      }
      scatter(nt0, acc0);
      scatter(nt0 + 1, acc1);
    }
  }
  __syncthreads();

  // ---- Final assembly: out = [out_s (32) | out_v (16x3)] * 1/sqrt(48) ----
  for (int idx = tid; idx < 16 * 80; idx += NBLK) {
    int e = idx / 80, o = idx - 80 * e;
    float res;
    if (o < 32) {
      res = INV_SQRT_FAN * outs_acc[e][o];
    } else {
      int q = o - 32;
      int u = q / 3, c = q - 3 * u;
      res = INV_SQRT_FAN * (sv_acc[e][u] * v2l[e][c] + vs_acc[e][u][c]);
    }
    out[(size_t)(eb + e) * 80 + o] = res;
  }
}

extern "C" void kernel_launch(void* const* d_in, const int* in_sizes, int n_in,
                              void* d_out, int out_size, void* d_ws, size_t ws_size,
                              hipStream_t stream) {
  const float* attr1     = (const float*)d_in[0];  // E x 80
  const float* attr2     = (const float*)d_in[1];  // E x 4
  const float* edge_attr = (const float*)d_in[2];  // E x 128
  const float* ln_g      = (const float*)d_in[3];
  const float* ln_b      = (const float*)d_in[4];
  const float* W1        = (const float*)d_in[5];  // 128 x 128
  const float* b1        = (const float*)d_in[6];
  const float* W2        = (const float*)d_in[7];  // 128 x 2304
  const float* b2        = (const float*)d_in[8];
  float* out             = (float*)d_out;

  const int E = in_sizes[1] / 4;                   // attr2 is E x 4

  _Float16* W1h = (_Float16*)d_ws;                 // 16384 halves (32 KB)
  _Float16* W2h = W1h + 16384;                     // 294912 halves (576 KB)

  const int total_halves = 16384 + 294912;         // 311296 -> 1216 blocks
  swizzle_weights<<<(total_halves + NBLK - 1) / NBLK, NBLK, 0, stream>>>(
      W1, W2, W1h, W2h);

  tpconv_fused<<<E / TILE_E, NBLK, 0, stream>>>(
      attr1, attr2, edge_attr, ln_g, ln_b, W1h, b1, W2h, b2, out);
}
